// FDDNet_68418829025669
// MI455X (gfx1250) — matrix operand order repaired and checked by reading, not compiled, not run on hardware
//
#include <hip/hip_runtime.h>
#include <stdint.h>

#define ACT_NONE 0
#define ACT_RELU 1
#define ACT_SILU 2
#define ACT_SIG  3

typedef __attribute__((ext_vector_type(16))) __bf16       v16bf;
typedef __attribute__((ext_vector_type(8)))  float        v8f;
typedef __attribute__((ext_vector_type(4)))  unsigned int v4u;
typedef __attribute__((ext_vector_type(8)))  int          v8i;
typedef __attribute__((ext_vector_type(4)))  int          v4i;

__device__ __forceinline__ unsigned short f2bf_bits(float f) {
  unsigned u = __builtin_bit_cast(unsigned, f);
  unsigned r = (u + 0x7FFFu + ((u >> 16) & 1u)) >> 16;   // round-to-nearest-even
  return (unsigned short)r;
}
__device__ __forceinline__ __bf16 bits2bf(unsigned short b) {
  return __builtin_bit_cast(__bf16, b);
}
__device__ __forceinline__ float act_apply(float o, int act) {
  if (act == ACT_RELU)      o = fmaxf(o, 0.f);
  else if (act == ACT_SILU) o = o / (1.f + __expf(-o));
  else if (act == ACT_SIG)  o = 1.f / (1.f + __expf(-o));
  return o;
}

// ---------------------------------------------------------------------------
// fp32 -> bf16 weight packing. Stages each 16KB tile into LDS with the CDNA5
// Tensor Data Mover (1-D descriptor, OOB reads return zero), waits on
// TENSORcnt, then converts out of LDS.
// ---------------------------------------------------------------------------
#define PACK_TILE 4096

__global__ void __launch_bounds__(256)
k_pack_bf16(const float* __restrict__ src, unsigned short* __restrict__ dst, long long n) {
  __shared__ float tile[PACK_TILE];
  const long long base = (long long)blockIdx.x * PACK_TILE;
  if (base >= n) return;
  long long rem = n - base;
  const int cnt = rem < (long long)PACK_TILE ? (int)rem : PACK_TILE;

#if __has_builtin(__builtin_amdgcn_tensor_load_to_lds)
  if ((threadIdx.x >> 5) == 0) {           // one wave issues the TDM op
    unsigned long long gaddr = (unsigned long long)(const void*)(src + base);
    unsigned lds_addr = (unsigned)(unsigned long long)(size_t)&tile[0];
    v4u g0;
    g0[0] = 1u;                                            // count = 1 valid D#
    g0[1] = lds_addr;                                      // lds byte address
    g0[2] = (unsigned)(gaddr & 0xFFFFFFFFull);             // global_addr lo
    g0[3] = (unsigned)((gaddr >> 32) & 0x01FFFFFFull) | 0x80000000u; // hi | type=2
    unsigned td0 = (unsigned)cnt;                          // tensor_dim0 (OOB -> 0)
    v8i g1;
    g1[0] = (int)(2u << 16);                               // data_size = 4B
    g1[1] = (int)((td0 & 0xFFFFu) << 16);                  // tensor_dim0[15:0]
    g1[2] = (int)((td0 >> 16) | (1u << 16));               // dim0 hi | tensor_dim1=1
    g1[3] = (int)(((unsigned)PACK_TILE) << 16);            // tile_dim0
    g1[4] = 0;                                             // tile_dim1=0 (1-D)
    g1[5] = (int)PACK_TILE;                                // dim0 stride lo
    g1[6] = 0; g1[7] = 0;
    v4i g2 = {0, 0, 0, 0}, g3 = {0, 0, 0, 0};
#if defined(__clang_major__) && (__clang_major__ >= 23)
    v8i g4 = {0, 0, 0, 0, 0, 0, 0, 0};
    __builtin_amdgcn_tensor_load_to_lds(g0, g1, g2, g3, g4, 0);
#else
    __builtin_amdgcn_tensor_load_to_lds(g0, g1, g2, g3, 0);
#endif
#if __has_builtin(__builtin_amdgcn_s_wait_tensorcnt)
    __builtin_amdgcn_s_wait_tensorcnt(0);
#endif
  }
  __syncthreads();
  for (int i = threadIdx.x; i < cnt; i += 256)
    dst[base + i] = f2bf_bits(tile[i]);
#else
  for (int i = threadIdx.x; i < cnt; i += 256)
    dst[base + i] = f2bf_bits(src[base + i]);
#endif
}

// ---------------------------------------------------------------------------
// Fold BN into per-channel scale/bias:  scale = g/sqrt(v+eps),
// bias = b - m*scale (+ extra*scale for conv-bias folded through BN).
// ---------------------------------------------------------------------------
__global__ void k_bn_prep(const float* g, const float* b, const float* m, const float* v,
                          float eps, const float* extra, float* scale, float* bias, int C) {
  int c = blockIdx.x * blockDim.x + threadIdx.x;
  if (c >= C) return;
  float inv = g[c] * rsqrtf(v[c] + eps);
  scale[c] = inv;
  float bb = b[c] - m[c] * inv;
  if (extra) bb += extra[c] * inv;
  bias[c] = bb;
}

__global__ void k_bias_prep(const float* bsrc, float* scale, float* bias, int C) {
  int c = blockIdx.x * blockDim.x + threadIdx.x;
  if (c >= C) return;
  scale[c] = 1.f;
  bias[c]  = bsrc ? bsrc[c] : 0.f;
}

// ---------------------------------------------------------------------------
// Implicit-GEMM conv via V_WMMA_F32_16X16X32_BF16.
//   GEMM: M = Cout, N = B*Ho*Wo, K = Cin*k*k (zero padded to 32)
// One block = 4 waves; all waves share one LDS-staged 16x32 A (weight) tile
// per k-step; each wave owns a 16M x 32N tile (two B fragments + two f32
// accumulators reusing the same A fragment). Epilogue fuses BN scale/bias,
// activation, residual add, channel-offset concat write, transposed tokens.
// ---------------------------------------------------------------------------
__global__ void __launch_bounds__(128)
k_conv_wmma(const float* __restrict__ x, const unsigned short* __restrict__ wq,
            const float* __restrict__ scale, const float* __restrict__ bias,
            const float* __restrict__ resid, float* __restrict__ out,
            int B, int Cin, int H, int W, int Cout, int Kk, int stride, int pad, int dil,
            int Ho, int Wo, int act,
            int in_cs, int in_co, int out_cs, int out_co,
            int tmode, int tok0) {
  __shared__ unsigned short As[16 * 32];

  const int lane = threadIdx.x & 31;
  const int wv   = threadIdx.x >> 5;
  const int m0   = blockIdx.x * 16;
  const int Kdim = Cin * Kk * Kk;
  const long long N  = (long long)B * Ho * Wo;
  const long long n0 = ((long long)blockIdx.y * 4 + wv) * 32;

  // two 16-pixel N tiles per wave
  const int nn = lane & 15;
  bool val_t[2]; int b_t[2], oy_t[2], ox_t[2];
#pragma unroll
  for (int t = 0; t < 2; ++t) {
    long long n = n0 + t * 16 + nn;
    val_t[t] = (n < N);
    int b = 0, oy = 0, ox = 0;
    if (val_t[t]) {
      long long q = n;
      b = (int)(q / ((long long)Ho * Wo));
      q -= (long long)b * Ho * Wo;
      oy = (int)(q / Wo);
      ox = (int)(q - (long long)oy * Wo);
    }
    b_t[t] = b; oy_t[t] = oy; ox_t[t] = ox;
  }
  const int kbA = (lane < 16) ? 0 : 8;    // A-frag K interleave (16-bit layout)
  const int kbB = (lane < 16) ? 0 : 16;   // unused: B-frag uses same K interleave as A
  const int am  = lane & 15;
  const int KK2 = Kk * Kk;

  v8f acc0 = {0.f, 0.f, 0.f, 0.f, 0.f, 0.f, 0.f, 0.f};
  v8f acc1 = acc0;

  for (int kk = 0; kk < Kdim; kk += 32) {
    // cooperative stage of A tile (16 x 32 bf16) into LDS
    for (int i = threadIdx.x; i < 512; i += 128) {
      int mm = i >> 5, kq = i & 31;
      unsigned short wb = 0;
      int mrow = m0 + mm, kidx = kk + kq;
      if (mrow < Cout && kidx < Kdim) wb = wq[(long long)mrow * Kdim + kidx];
      As[(mm << 5) + kq] = wb;
    }
    // prefetch next weight k-slice while this step computes
    if (kk + 32 < Kdim)
      __builtin_prefetch(&wq[(long long)(m0 + (threadIdx.x & 15)) * Kdim + kk + 32], 0, 1);
    __syncthreads();

    v16bf a;
#pragma unroll
    for (int v = 0; v < 8; ++v) {
      const int kA = (v < 4) ? (kbA + 2 * v) : (16 + kbA + 2 * (v - 4));
      a[2 * v]     = bits2bf(As[(am << 5) + kA]);
      a[2 * v + 1] = bits2bf(As[(am << 5) + kA + 1]);
    }

    v16bf bm_t[2];
#pragma unroll
    for (int t = 0; t < 2; ++t) {
      const bool nv = val_t[t];
      const int b = b_t[t], oy = oy_t[t], ox = ox_t[t];
#pragma unroll
      for (int v = 0; v < 8; ++v) {
        const int kB = (v < 4) ? (kbA + 2 * v) : (16 + kbA + 2 * (v - 4));
#pragma unroll
        for (int e = 0; e < 2; ++e) {
          int kidx = kk + kB + e;
          float val = 0.f;
          if (nv && kidx < Kdim) {
            int ci = kidx / KK2;
            int r  = kidx - ci * KK2;
            int ky = r / Kk, kx = r - ky * Kk;
            int iy = oy * stride - pad + ky * dil;
            int ix = ox * stride - pad + kx * dil;
            if (iy >= 0 && iy < H && ix >= 0 && ix < W)
              val = x[(((long long)b * in_cs + in_co + ci) * H + iy) * W + ix];
          }
          bm_t[t][2 * v + e] = bits2bf(f2bf_bits(val));
        }
      }
    }
    acc0 = __builtin_amdgcn_wmma_f32_16x16x32_bf16(false, a, false, bm_t[0],
                                                   (short)0, acc0, false, false);
    acc1 = __builtin_amdgcn_wmma_f32_16x16x32_bf16(false, a, false, bm_t[1],
                                                   (short)0, acc1, false, false);
    __syncthreads();
  }

  // epilogue
  const int mb = (lane < 16) ? 0 : 8;
#pragma unroll
  for (int t = 0; t < 2; ++t) {
    if (!val_t[t]) continue;
    const int b = b_t[t], oy = oy_t[t], ox = ox_t[t];
#pragma unroll
    for (int v = 0; v < 8; ++v) {
      int m = m0 + mb + v;
      if (m >= Cout) continue;
      float o = (t == 0 ? acc0[v] : acc1[v]) * scale[m] + bias[m];
      o = act_apply(o, act);
      if (resid) o += resid[(((long long)b * Cout + m) * Ho + oy) * Wo + ox];
      if (!tmode) {
        out[(((long long)b * out_cs + out_co + m) * Ho + oy) * Wo + ox] = o;
      } else {  // token-transposed projection write: [B, 8400, 224]
        long long pix = (long long)oy * Wo + ox;
        out[((long long)b * 8400 + tok0 + pix) * 224 + out_co + m] = o;
      }
    }
  }
}

// ---------------------------------------------------------------------------
// Specialized 1x1 conv (pure GEMM): the K index *is* the input channel, so
// the B gather is one indexed load off a per-lane base pointer. This carries
// most of the network's FLOPs (c2f/cv4/SPPF/deform-einsum/projection heads).
// ---------------------------------------------------------------------------
__global__ void __launch_bounds__(128)
k_conv_wmma_1x1(const float* __restrict__ x, const unsigned short* __restrict__ wq,
                const float* __restrict__ scale, const float* __restrict__ bias,
                const float* __restrict__ resid, float* __restrict__ out,
                int B, int Cin, int H, int W, int Cout, int act,
                int in_cs, int in_co, int out_cs, int out_co,
                int tmode, int tok0) {
  __shared__ unsigned short As[16 * 32];

  const int lane = threadIdx.x & 31;
  const int wv   = threadIdx.x >> 5;
  const int m0   = blockIdx.x * 16;
  const long long HWsz = (long long)H * W;
  const long long N  = (long long)B * HWsz;
  const long long n0 = ((long long)blockIdx.y * 4 + wv) * 32;

  const int nn = lane & 15;
  bool val_t[2]; int b_t[2], oy_t[2], ox_t[2];
  const float* px_t[2];
#pragma unroll
  for (int t = 0; t < 2; ++t) {
    long long n = n0 + t * 16 + nn;
    val_t[t] = (n < N);
    int b = 0, oy = 0, ox = 0;
    if (val_t[t]) {
      long long q = n;
      b = (int)(q / HWsz);
      q -= (long long)b * HWsz;
      oy = (int)(q / W);
      ox = (int)(q - (long long)oy * W);
    }
    b_t[t] = b; oy_t[t] = oy; ox_t[t] = ox;
    px_t[t] = x + (((long long)b * in_cs + in_co) * H + oy) * W + ox;
  }
  const int kbA = (lane < 16) ? 0 : 8;
  const int kbB = (lane < 16) ? 0 : 16;
  const int am  = lane & 15;

  v8f acc0 = {0.f, 0.f, 0.f, 0.f, 0.f, 0.f, 0.f, 0.f};
  v8f acc1 = acc0;

  for (int kk = 0; kk < Cin; kk += 32) {
    for (int i = threadIdx.x; i < 512; i += 128) {
      int mm = i >> 5, kq = i & 31;
      unsigned short wb = 0;
      int mrow = m0 + mm, kidx = kk + kq;
      if (mrow < Cout && kidx < Cin) wb = wq[(long long)mrow * Cin + kidx];
      As[(mm << 5) + kq] = wb;
    }
    if (kk + 32 < Cin)
      __builtin_prefetch(&wq[(long long)(m0 + (threadIdx.x & 15)) * Cin + kk + 32], 0, 1);
    __syncthreads();

    v16bf a;
#pragma unroll
    for (int v = 0; v < 8; ++v) {
      const int kA = (v < 4) ? (kbA + 2 * v) : (16 + kbA + 2 * (v - 4));
      a[2 * v]     = bits2bf(As[(am << 5) + kA]);
      a[2 * v + 1] = bits2bf(As[(am << 5) + kA + 1]);
    }

    v16bf bm_t[2];
#pragma unroll
    for (int t = 0; t < 2; ++t) {
      const bool nv = val_t[t];
      const float* px = px_t[t];
#pragma unroll
      for (int v = 0; v < 8; ++v) {
        const int kB = (v < 4) ? (kbA + 2 * v) : (16 + kbA + 2 * (v - 4));
#pragma unroll
        for (int e = 0; e < 2; ++e) {
          int ci = kk + kB + e;
          float val = (nv && ci < Cin) ? px[(long long)ci * HWsz] : 0.f;
          bm_t[t][2 * v + e] = bits2bf(f2bf_bits(val));
        }
      }
    }
    acc0 = __builtin_amdgcn_wmma_f32_16x16x32_bf16(false, a, false, bm_t[0],
                                                   (short)0, acc0, false, false);
    acc1 = __builtin_amdgcn_wmma_f32_16x16x32_bf16(false, a, false, bm_t[1],
                                                   (short)0, acc1, false, false);
    __syncthreads();
  }

  const int mb = (lane < 16) ? 0 : 8;
#pragma unroll
  for (int t = 0; t < 2; ++t) {
    if (!val_t[t]) continue;
    const int b = b_t[t], oy = oy_t[t], ox = ox_t[t];
#pragma unroll
    for (int v = 0; v < 8; ++v) {
      int m = m0 + mb + v;
      if (m >= Cout) continue;
      float o = (t == 0 ? acc0[v] : acc1[v]) * scale[m] + bias[m];
      o = act_apply(o, act);
      if (resid) o += resid[(((long long)b * Cout + m) * H + oy) * W + ox];
      if (!tmode) {
        out[(((long long)b * out_cs + out_co + m) * H + oy) * W + ox] = o;
      } else {
        long long pix = (long long)oy * W + ox;
        out[((long long)b * 8400 + tok0 + pix) * 224 + out_co + m] = o;
      }
    }
  }
}

// ---------------------------------------------------------------------------
// glue kernels
// ---------------------------------------------------------------------------
__global__ void k_maxpool5(const float* __restrict__ in, float* __restrict__ out,
                           int B, int C, int H, int W, int cs, int ci, int co) {
  long long i = (long long)blockIdx.x * blockDim.x + threadIdx.x;
  long long tot = (long long)B * C * H * W;
  if (i >= tot) return;
  int xx = (int)(i % W); long long t = i / W;
  int yy = (int)(t % H); t /= H;
  int c  = (int)(t % C); int b = (int)(t / C);
  float mx = -INFINITY;
  for (int dy = -2; dy <= 2; ++dy) {
    int y = yy + dy; if (y < 0 || y >= H) continue;
    for (int dx = -2; dx <= 2; ++dx) {
      int x = xx + dx; if (x < 0 || x >= W) continue;
      mx = fmaxf(mx, in[(((long long)b * cs + ci + c) * H + y) * W + x]);
    }
  }
  out[(((long long)b * cs + co + c) * H + yy) * W + xx] = mx;
}

__global__ void k_up2(const float* __restrict__ in, float* __restrict__ out,
                      int B, int C, int H, int W, int ics, int ico, int ocs, int oco) {
  const int Ho = 2 * H, Wo = 2 * W;
  long long i = (long long)blockIdx.x * blockDim.x + threadIdx.x;
  long long tot = (long long)B * C * Ho * Wo;
  if (i >= tot) return;
  int x = (int)(i % Wo); long long t = i / Wo;
  int y = (int)(t % Ho); t /= Ho;
  int c = (int)(t % C);  int b = (int)(t / C);
  out[(((long long)b * ocs + oco + c) * Ho + y) * Wo + x] =
      in[(((long long)b * ics + ico + c) * H + (y >> 1)) * W + (x >> 1)];
}

__global__ void k_copy_ch(const float* __restrict__ in, float* __restrict__ out,
                          int B, int C, int H, int W, int ics, int ico, int ocs, int oco) {
  long long i = (long long)blockIdx.x * blockDim.x + threadIdx.x;
  long long tot = (long long)B * C * H * W;
  if (i >= tot) return;
  int x = (int)(i % W); long long t = i / W;
  int y = (int)(t % H); t /= H;
  int c = (int)(t % C); int b = (int)(t / C);
  out[(((long long)b * ocs + oco + c) * H + y) * W + x] =
      in[(((long long)b * ics + ico + c) * H + y) * W + x];
}

// Modulated deformable sampling: produces val[b, ci*9+k, ho, wo] so that the
// einsum 'bkchw,ock->bohw' becomes a 1x1 WMMA conv with dw viewed [Cout,Cin*9].
__global__ void k_deform_gather(const float* __restrict__ x, const float* __restrict__ off,
                                const float* __restrict__ msk, float* __restrict__ val,
                                int B, int Cin, int H, int W, int Ho, int Wo, int stride) {
  long long i = (long long)blockIdx.x * blockDim.x + threadIdx.x;
  long long tot = (long long)B * 9 * Ho * Wo;
  if (i >= tot) return;
  int wo = (int)(i % Wo); long long t = i / Wo;
  int ho = (int)(t % Ho); t /= Ho;
  int k  = (int)(t % 9);  int b = (int)(t / 9);

  float dy = off[(((long long)b * 18 + 2 * k) * Ho + ho) * Wo + wo];
  float dx = off[(((long long)b * 18 + 2 * k + 1) * Ho + ho) * Wo + wo];
  float mv = msk[(((long long)b * 9 + k) * Ho + ho) * Wo + wo];

  float py = (float)(ho * stride - 1 + (k / 3)) + dy;
  float px = (float)(wo * stride - 1 + (k % 3)) + dx;
  float y0f = floorf(py), x0f = floorf(px);
  float wy = py - y0f, wx = px - x0f;
  int y0 = (int)y0f, x0 = (int)x0f;
  int y1 = y0 + 1, x1 = x0 + 1;
  bool vy0 = (y0 >= 0) && (y0 < H), vy1 = (y1 >= 0) && (y1 < H);
  bool vx0 = (x0 >= 0) && (x0 < W), vx1 = (x1 >= 0) && (x1 < W);
  float w00 = (1.f - wy) * (1.f - wx) * (vy0 && vx0 ? 1.f : 0.f);
  float w01 = (1.f - wy) * wx         * (vy0 && vx1 ? 1.f : 0.f);
  float w10 = wy * (1.f - wx)         * (vy1 && vx0 ? 1.f : 0.f);
  float w11 = wy * wx                 * (vy1 && vx1 ? 1.f : 0.f);
  int cy0 = y0 < 0 ? 0 : (y0 >= H ? H - 1 : y0);
  int cy1 = y1 < 0 ? 0 : (y1 >= H ? H - 1 : y1);
  int cx0 = x0 < 0 ? 0 : (x0 >= W ? W - 1 : x0);
  int cx1 = x1 < 0 ? 0 : (x1 >= W ? W - 1 : x1);

  for (int ci = 0; ci < Cin; ++ci) {
    const float* xp = x + ((long long)b * Cin + ci) * H * W;
    float s = w00 * xp[cy0 * W + cx0] + w01 * xp[cy0 * W + cx1] +
              w10 * xp[cy1 * W + cx0] + w11 * xp[cy1 * W + cx1];
    val[(((long long)b * Cin * 9 + ci * 9 + k) * Ho + ho) * Wo + wo] = s * mv;
  }
}

// ---------------------------------------------------------------------------
// host orchestration
// ---------------------------------------------------------------------------
extern "C" void kernel_launch(void* const* d_in, const int* in_sizes, int n_in,
                              void* d_out, int out_size, void* d_ws, size_t ws_size,
                              hipStream_t stream) {
  (void)in_sizes; (void)n_in; (void)out_size; (void)ws_size;
  const int B = 2;
  int cur = 0;
  auto nxt = [&]() -> const float* { return (const float*)d_in[cur++]; };

  struct Bn { const float *g, *b, *m, *v; };
  struct Cv { const float* w; Bn bn; };
  struct C2f { Cv cv1, m1, m2, cv2; };
  struct Sfab { const float* w1; Bn bn1; Cv bcv1, bcv2; const float* w2; Bn bn2;
                const float *dw, *db, *ow, *ob, *mw, *mb; Bn bn3; Cv cv4; };
  struct Proj { const float* w; Bn bn; };

  auto pBn  = [&]() { Bn r; r.g = nxt(); r.b = nxt(); r.m = nxt(); r.v = nxt(); return r; };
  auto pCv  = [&]() { Cv c; c.w = nxt(); c.bn = pBn(); return c; };
  auto pC2f = [&]() { C2f c; c.cv1 = pCv(); c.m1 = pCv(); c.m2 = pCv(); c.cv2 = pCv(); return c; };

  // setup_inputs() insertion order: x, layers[5], sppf, fpn{up1,up2,ds1,dn1,ds2,dn2}, proj[3]
  const float* x = nxt();
  Sfab L[5];
  for (int i = 0; i < 5; ++i) {
    Sfab s; s.w1 = nxt(); s.bn1 = pBn(); s.bcv1 = pCv(); s.bcv2 = pCv();
    s.w2 = nxt(); s.bn2 = pBn(); s.dw = nxt(); s.db = nxt(); s.ow = nxt(); s.ob = nxt();
    s.mw = nxt(); s.mb = nxt(); s.bn3 = pBn(); s.cv4 = pCv(); L[i] = s;
  }
  Cv sp1 = pCv(), sp2 = pCv();
  C2f up1 = pC2f(), up2c = pC2f(); Cv ds1 = pCv(); C2f dn1 = pC2f(); Cv ds2 = pCv(); C2f dn2 = pC2f();
  Proj pr[3];
  for (int i = 0; i < 3; ++i) { pr[i].w = nxt(); pr[i].bn = pBn(); }

  // bump allocator over d_ws (stream ordering makes mark/reset reuse safe)
  char* wsb = (char*)d_ws; size_t wso = 0;
  auto alloc  = [&](size_t bytes) -> void* {
    size_t o = (wso + 255) & ~(size_t)255; wso = o + bytes; return wsb + o;
  };
  auto allocf = [&](long long elems) -> float* { return (float*)alloc((size_t)elems * 4); };

  auto launch_conv = [&](const float* in, const float* w,
                         int cin, int cout, int k, int H, int W, int stride, int pad, int dil,
                         const Bn* bn, float eps, const float* biasp, const float* extra,
                         int act, const float* resid, float* out,
                         int in_cs, int in_co, int out_cs, int out_co,
                         int tmode, int tok0) {
    int eff = dil * (k - 1) + 1;
    int Ho = (H + 2 * pad - eff) / stride + 1;
    int Wo = (W + 2 * pad - eff) / stride + 1;
    long long Kd = (long long)cin * k * k;
    long long nw = (long long)cout * Kd;
    unsigned short* wq = (unsigned short*)alloc((size_t)nw * 2);
    float* sb = allocf(2LL * cout);
    k_pack_bf16<<<dim3((unsigned)((nw + PACK_TILE - 1) / PACK_TILE)), dim3(256), 0, stream>>>(w, wq, nw);
    if (bn)
      k_bn_prep<<<dim3((cout + 63) / 64), dim3(64), 0, stream>>>(bn->g, bn->b, bn->m, bn->v,
                                                                 eps, extra, sb, sb + cout, cout);
    else
      k_bias_prep<<<dim3((cout + 63) / 64), dim3(64), 0, stream>>>(biasp, sb, sb + cout, cout);
    long long N = (long long)B * Ho * Wo;
    dim3 grid((unsigned)((cout + 15) / 16), (unsigned)((N + 127) / 128));
    if (k == 1 && stride == 1 && pad == 0)
      k_conv_wmma_1x1<<<grid, dim3(128), 0, stream>>>(in, wq, sb, sb + cout, resid, out,
                                                      B, cin, H, W, cout, act,
                                                      in_cs, in_co, out_cs, out_co, tmode, tok0);
    else
      k_conv_wmma<<<grid, dim3(128), 0, stream>>>(in, wq, sb, sb + cout, resid, out,
                                                  B, cin, H, W, cout, k, stride, pad, dil,
                                                  Ho, Wo, act, in_cs, in_co, out_cs, out_co,
                                                  tmode, tok0);
  };

  auto run_sfab = [&](const Sfab& s, const float* in, int cin, int cout, int H, float* outb) {
    int Ho = H / 2;
    size_t mark = wso;
    long long A2 = (long long)B * Ho * Ho;
    float* cat = allocf((long long)(3 * cout) * A2);
    float* x1  = allocf((long long)cout * A2);
    float* tb  = allocf((long long)(cout / 2) * A2);
    float* ofb = allocf(18LL * A2);
    float* msb = allocf(9LL * A2);
    float* vb  = allocf((long long)(9 * cin) * A2);
    // x1 = relu(bn1(conv3x3 s2))
    launch_conv(in, s.w1, cin, cout, 3, H, H, 2, 1, 1, &s.bn1, 1e-5f, nullptr, nullptr,
                ACT_RELU, nullptr, x1, cin, 0, cout, 0, 0, 0);
    // bottleneck e=0.5 with shortcut -> cat[0:cout]
    launch_conv(x1, s.bcv1.w, cout, cout / 2, 3, Ho, Ho, 1, 1, 1, &s.bcv1.bn, 1e-3f, nullptr, nullptr,
                ACT_SILU, nullptr, tb, cout, 0, cout / 2, 0, 0, 0);
    launch_conv(tb, s.bcv2.w, cout / 2, cout, 3, Ho, Ho, 1, 1, 1, &s.bcv2.bn, 1e-3f, nullptr, nullptr,
                ACT_SILU, x1, cat, cout / 2, 0, 3 * cout, 0, 0, 0);
    // x2 = relu(bn2(conv3x3 s2 d2)) -> cat[cout:2cout]
    launch_conv(in, s.w2, cin, cout, 3, H, H, 2, 2, 2, &s.bn2, 1e-5f, nullptr, nullptr,
                ACT_RELU, nullptr, cat, cin, 0, 3 * cout, cout, 0, 0);
    // deformable branch -> cat[2cout:3cout]
    launch_conv(in, s.ow, cin, 18, 3, H, H, 2, 1, 1, nullptr, 0.f, s.ob, nullptr,
                ACT_NONE, nullptr, ofb, cin, 0, 18, 0, 0, 0);
    launch_conv(in, s.mw, cin, 9, 3, H, H, 2, 1, 1, nullptr, 0.f, s.mb, nullptr,
                ACT_SIG, nullptr, msb, cin, 0, 9, 0, 0, 0);
    long long tg = (long long)B * 9 * Ho * Ho;
    k_deform_gather<<<dim3((unsigned)((tg + 255) / 256)), dim3(256), 0, stream>>>(
        in, ofb, msb, vb, B, cin, H, H, Ho, Ho, 2);
    launch_conv(vb, s.dw, 9 * cin, cout, 1, Ho, Ho, 1, 0, 1, &s.bn3, 1e-5f, nullptr, s.db,
                ACT_RELU, nullptr, cat, 9 * cin, 0, 3 * cout, 2 * cout, 0, 0);
    // cv4: 1x1 fuse
    launch_conv(cat, s.cv4.w, 3 * cout, cout, 1, Ho, Ho, 1, 0, 1, &s.cv4.bn, 1e-3f, nullptr, nullptr,
                ACT_SILU, nullptr, outb, 3 * cout, 0, cout, 0, 0, 0);
    wso = mark;
  };

  auto run_c2f = [&](const C2f& c, const float* in, int cin, int cout, int H, float* outb) {
    int cc = cout / 2;
    size_t mark = wso;
    long long A2 = (long long)B * H * H;
    float* cat = allocf((long long)(3 * cc) * A2);
    float* tb  = allocf((long long)cc * A2);
    launch_conv(in, c.cv1.w, cin, 2 * cc, 1, H, H, 1, 0, 1, &c.cv1.bn, 1e-3f, nullptr, nullptr,
                ACT_SILU, nullptr, cat, cin, 0, 3 * cc, 0, 0, 0);
    launch_conv(cat, c.m1.w, cc, cc, 3, H, H, 1, 1, 1, &c.m1.bn, 1e-3f, nullptr, nullptr,
                ACT_SILU, nullptr, tb, 3 * cc, cc, cc, 0, 0, 0);
    launch_conv(tb, c.m2.w, cc, cc, 3, H, H, 1, 1, 1, &c.m2.bn, 1e-3f, nullptr, nullptr,
                ACT_SILU, nullptr, cat, cc, 0, 3 * cc, 2 * cc, 0, 0);
    launch_conv(cat, c.cv2.w, 3 * cc, cout, 1, H, H, 1, 0, 1, &c.cv2.bn, 1e-3f, nullptr, nullptr,
                ACT_SILU, nullptr, outb, 3 * cc, 0, cout, 0, 0, 0);
    wso = mark;
  };

  // persistent feature buffers
  float* l1  = allocf(2LL * 14 * 320 * 320);
  float* l2  = allocf(2LL * 28 * 160 * 160);
  float* f1  = allocf(2LL * 56 * 80 * 80);
  float* f2  = allocf(2LL * 112 * 40 * 40);
  float* f3r = allocf(2LL * 224 * 20 * 20);
  float* f3  = allocf(2LL * 224 * 20 * 20);
  float* P4  = allocf(2LL * 112 * 40 * 40);
  float* P3  = allocf(2LL * 56 * 80 * 80);
  float* P4b = allocf(2LL * 112 * 40 * 40);
  float* P5  = allocf(2LL * 224 * 20 * 20);

  run_sfab(L[0], x,   10,  14, 640, l1);
  run_sfab(L[1], l1,  14,  28, 320, l2);
  run_sfab(L[2], l2,  28,  56, 160, f1);
  run_sfab(L[3], f1,  56, 112,  80, f2);
  run_sfab(L[4], f2, 112, 224,  40, f3r);

  { // SPPF
    size_t mark = wso;
    float* cat = allocf(2LL * 448 * 20 * 20);
    launch_conv(f3r, sp1.w, 224, 112, 1, 20, 20, 1, 0, 1, &sp1.bn, 1e-3f, nullptr, nullptr,
                ACT_SILU, nullptr, cat, 224, 0, 448, 0, 0, 0);
    long long tot = 2LL * 112 * 20 * 20;
    for (int i = 0; i < 3; ++i)
      k_maxpool5<<<dim3((unsigned)((tot + 255) / 256)), dim3(256), 0, stream>>>(
          cat, cat, B, 112, 20, 20, 448, i * 112, (i + 1) * 112);
    launch_conv(cat, sp2.w, 448, 224, 1, 20, 20, 1, 0, 1, &sp2.bn, 1e-3f, nullptr, nullptr,
                ACT_SILU, nullptr, f3, 448, 0, 224, 0, 0, 0);
    wso = mark;
  }
  { // FPN top-down 1: cat(up2(feat3), feat2) -> c2f -> P4
    size_t mark = wso;
    float* cat = allocf(2LL * 336 * 40 * 40);
    long long t1 = 2LL * 224 * 40 * 40, t2 = 2LL * 112 * 40 * 40;
    k_up2<<<dim3((unsigned)((t1 + 255) / 256)), dim3(256), 0, stream>>>(f3, cat, B, 224, 20, 20, 224, 0, 336, 0);
    k_copy_ch<<<dim3((unsigned)((t2 + 255) / 256)), dim3(256), 0, stream>>>(f2, cat, B, 112, 40, 40, 112, 0, 336, 224);
    run_c2f(up1, cat, 336, 112, 40, P4);
    wso = mark;
  }
  { // FPN top-down 2: cat(up2(P4), feat1) -> c2f -> P3
    size_t mark = wso;
    float* cat = allocf(2LL * 168 * 80 * 80);
    long long t1 = 2LL * 112 * 80 * 80, t2 = 2LL * 56 * 80 * 80;
    k_up2<<<dim3((unsigned)((t1 + 255) / 256)), dim3(256), 0, stream>>>(P4, cat, B, 112, 40, 40, 112, 0, 168, 0);
    k_copy_ch<<<dim3((unsigned)((t2 + 255) / 256)), dim3(256), 0, stream>>>(f1, cat, B, 56, 80, 80, 56, 0, 168, 112);
    run_c2f(up2c, cat, 168, 56, 80, P3);
    wso = mark;
  }
  { // bottom-up 1: cat(ds1(P3), P4) -> c2f -> P4b
    size_t mark = wso;
    float* cat = allocf(2LL * 168 * 40 * 40);
    launch_conv(P3, ds1.w, 56, 56, 3, 80, 80, 2, 1, 1, &ds1.bn, 1e-3f, nullptr, nullptr,
                ACT_SILU, nullptr, cat, 56, 0, 168, 0, 0, 0);
    long long t2 = 2LL * 112 * 40 * 40;
    k_copy_ch<<<dim3((unsigned)((t2 + 255) / 256)), dim3(256), 0, stream>>>(P4, cat, B, 112, 40, 40, 112, 0, 168, 56);
    run_c2f(dn1, cat, 168, 112, 40, P4b);
    wso = mark;
  }
  { // bottom-up 2: cat(ds2(P4b), feat3) -> c2f -> P5
    size_t mark = wso;
    float* cat = allocf(2LL * 336 * 20 * 20);
    launch_conv(P4b, ds2.w, 112, 112, 3, 40, 40, 2, 1, 1, &ds2.bn, 1e-3f, nullptr, nullptr,
                ACT_SILU, nullptr, cat, 112, 0, 336, 0, 0, 0);
    long long t2 = 2LL * 224 * 20 * 20;
    k_copy_ch<<<dim3((unsigned)((t2 + 255) / 256)), dim3(256), 0, stream>>>(f3, cat, B, 224, 20, 20, 224, 0, 336, 112);
    run_c2f(dn2, cat, 336, 224, 20, P5);
    wso = mark;
  }

  // projection heads: 1x1 conv + BN, written token-transposed into [B, 8400, 224]
  float* out = (float*)d_out;
  launch_conv(P3,  pr[0].w,  56, 224, 1, 80, 80, 1, 0, 1, &pr[0].bn, 1e-5f, nullptr, nullptr,
              ACT_NONE, nullptr, out,  56, 0, 224, 0, 1, 0);
  launch_conv(P4b, pr[1].w, 112, 224, 1, 40, 40, 1, 0, 1, &pr[1].bn, 1e-5f, nullptr, nullptr,
              ACT_NONE, nullptr, out, 112, 0, 224, 0, 1, 6400);
  launch_conv(P5,  pr[2].w, 224, 224, 1, 20, 20, 1, 0, 1, &pr[2].bn, 1e-5f, nullptr, nullptr,
              ACT_NONE, nullptr, out, 224, 0, 224, 0, 1, 8000);
}
